// EdgeFeatures_48601849922135
// MI455X (gfx1250) — matrix operand order repaired and checked by reading, not compiled, not run on hardware
//
#include <hip/hip_runtime.h>

// Problem dims (compile-time constants from the reference)
#define Bb 4
#define Nn 200
#define Hh 300
#define KP 320          // K padded to multiple of 32 (data region)
#define KSTRIDE 328     // LDS row stride in bf16: 164 dwords -> conflict-free b128 reads
#define GP 320          // output cols padded to 20 tiles (last tile fully masked)
#define NPAIRS 10       // 20 N-tiles processed 2 at a time
#define KSTEPS 10
#define WAVES 8         // waves per workgroup
#define MT 16           // M tile (rows per wave)

typedef __attribute__((ext_vector_type(16))) __bf16       v16bf;
typedef __attribute__((ext_vector_type(8)))  float        v8f;
typedef __attribute__((ext_vector_type(4)))  float        f32x4;
typedef __attribute__((ext_vector_type(4)))  unsigned int u32x4;
typedef __attribute__((ext_vector_type(2)))  unsigned int u32x2;

union V16U { u32x4 q[2]; v16bf v; };
union PK2  { __bf16 h[2]; unsigned int u; };

__device__ __forceinline__ unsigned int pack_bf16(float a, float b) {
  PK2 pk;
  pk.h[0] = (__bf16)a;
  pk.h[1] = (__bf16)b;
  return pk.u;
}

// A fragment (16x32 bf16, MxK): lane half h, elems 0-7 = K[h*8..h*8+7],
// elems 8-15 = K[16+h*8 .. 16+h*8+7]  -> two b128 loads 32 bytes apart.
__device__ __forceinline__ v16bf load_frag_a(const __bf16* p) {
  V16U t;
  t.q[0] = *(const u32x4*)(p);
  t.q[1] = *(const u32x4*)(p + 16);
  return t.v;
}
// B fragment (32x16 bf16, KxN): lane half h, elems 0-7 = K[h*8..h*8+7], elems 8-15 = K[16+h*8 .. 16+h*8+7].
__device__ __forceinline__ v16bf load_frag_b(const __bf16* p) {
  V16U t;
  t.q[0] = *(const u32x4*)(p);
  t.q[1] = *(const u32x4*)(p + 16);
  return t.v;
}

// ---------------------------------------------------------------------------
// Kernel 1: Vx'[b,n,g] = sum_h x[b,n,h]*V_w[g,h] + V_b[g] + 0.5*U_b[g]
// (U_b folded in so the main epilogue is just Vx'_i + Vx'_j). 144 MFLOP total.
// ---------------------------------------------------------------------------
__global__ void vx_bias_kernel(const float* __restrict__ x,
                               const float* __restrict__ V_w,
                               const float* __restrict__ V_b,
                               const float* __restrict__ U_b,
                               float* __restrict__ Vx) {
  __shared__ float xr[Hh];
  const int bn = blockIdx.x;
  for (int h = threadIdx.x; h < Hh; h += blockDim.x) xr[h] = x[bn * Hh + h];
  __syncthreads();
  for (int g = threadIdx.x; g < Hh; g += blockDim.x) {
    const float* wrow = V_w + g * Hh;
    float acc = 0.f;
#pragma unroll 4
    for (int h = 0; h < Hh; ++h) acc += xr[h] * wrow[h];
    Vx[bn * Hh + g] = acc + V_b[g] + 0.5f * U_b[g];
  }
}

// ---------------------------------------------------------------------------
// Kernel 2: main GEMM  out[r,g] = sum_h e[r,h]*U_w[g,h] + Vx'[q(r),g] + Vx'[p(r),g]
// bf16 WMMA 16x16x32, f32 accumulate. U_w staged once per block in LDS (bf16),
// each wave stages its 16-row e tile in LDS (bf16). LDS rows padded to 328
// bf16 so fragment ds_load_b128s hit all 64 banks exactly once. Two N-tiles
// are processed per pass with independent accumulators so B-fragment ds loads
// overlap with WMMA execution.
// ---------------------------------------------------------------------------
__global__ void __launch_bounds__(WAVES * 32, 1)
edge_wmma_kernel(const float* __restrict__ e,
                 const float* __restrict__ U_w,
                 const float* __restrict__ Vx,
                 float* __restrict__ out) {
  extern __shared__ char smem[];
  __bf16* Bw = (__bf16*)smem;              // GP   rows x KSTRIDE
  __bf16* Ae = Bw + (size_t)GP * KSTRIDE;  // W*MT rows x KSTRIDE

  const int tid  = threadIdx.x;
  const int wave = tid >> 5;
  const int lane = tid & 31;

  // ---- stage B = U_w (row g = output col, contiguous in h) as bf16, zero-padded.
  // Quad-granularity: 4 f32 loads (b128) -> 2 packed dwords (ds_store_b64).
  for (int p = tid; p < GP * (KP / 4); p += WAVES * 32) {
    const int g  = p / (KP / 4);
    const int kq = (p - g * (KP / 4)) * 4;
    f32x4 f = {0.f, 0.f, 0.f, 0.f};
    if (g < Hh && kq < Hh) f = *(const f32x4*)(U_w + (size_t)g * Hh + kq);
    u32x2 pk;
    pk.x = pack_bf16(f.x, f.y);
    pk.y = pack_bf16(f.z, f.w);
    *(u32x2*)(Bw + (size_t)g * KSTRIDE + kq) = pk;
  }
  __syncthreads();

  const int tile  = blockIdx.x * WAVES + wave;
  const int Mbase = tile * MT;

  // ---- stage this wave's 16 rows of e as bf16 (K padded 300->320 with zeros)
  __bf16* Am = Ae + (size_t)wave * MT * KSTRIDE;
  for (int p = lane; p < MT * (KP / 4); p += 32) {
    const int m  = p / (KP / 4);
    const int kq = (p - m * (KP / 4)) * 4;
    f32x4 f = {0.f, 0.f, 0.f, 0.f};
    if (kq < Hh) f = *(const f32x4*)(e + (size_t)(Mbase + m) * Hh + kq);
    u32x2 pk;
    pk.x = pack_bf16(f.x, f.y);
    pk.y = pack_bf16(f.z, f.w);
    *(u32x2*)(Am + (size_t)m * KSTRIDE + kq) = pk;
  }
  // same-wave LDS write->read: DScnt ordering enforced by compiler waits

  const int half = lane >> 4;
  const int nl   = lane & 15;

  // ---- decode the 8 output rows this lane owns: row = (b*N+i)*N + j
  int vxiB[8], vxjB[8];
#pragma unroll
  for (int r = 0; r < 8; ++r) {
    const int row = Mbase + half * 8 + r;
    const int q   = row / Nn;  // = b*N + i
    const int j   = row - q * Nn;
    const int b   = q / Nn;
    vxiB[r] = q * Hh;
    vxjB[r] = (b * Nn + j) * Hh;
  }

  // ---- hoist all 10 A fragments (80 VGPRs), reused across all N tiles
  const __bf16* Arow = Am + (size_t)nl * KSTRIDE + half * 8;
  v16bf av[KSTEPS];
#pragma unroll
  for (int ks = 0; ks < KSTEPS; ++ks) av[ks] = load_frag_a(Arow + ks * 32);

  for (int ntp = 0; ntp < NPAIRS; ++ntp) {
    const __bf16* Brow0 = Bw + (size_t)(ntp * 32 + nl) * KSTRIDE + half * 8;
    const __bf16* Brow1 = Brow0 + (size_t)16 * KSTRIDE;
    v8f acc0 = {0.f, 0.f, 0.f, 0.f, 0.f, 0.f, 0.f, 0.f};
    v8f acc1 = {0.f, 0.f, 0.f, 0.f, 0.f, 0.f, 0.f, 0.f};

    // two independent dep chains per step: wmma(acc0) runs while b1 loads land
#pragma unroll
    for (int ks = 0; ks < KSTEPS; ++ks) {
      v16bf b0 = load_frag_b(Brow0 + ks * 32);
      v16bf b1 = load_frag_b(Brow1 + ks * 32);
      acc0 = __builtin_amdgcn_wmma_f32_16x16x32_bf16(
          false, av[ks], false, b0, (short)0, acc0, false, false);
      acc1 = __builtin_amdgcn_wmma_f32_16x16x32_bf16(
          false, av[ks], false, b1, (short)0, acc1, false, false);
    }

    // epilogue tile 0 (g in [32*ntp, 32*ntp+16))
    {
      const int g = ntp * 32 + nl;
      if (g < Hh) {
#pragma unroll
        for (int r = 0; r < 8; ++r) {
          const float v = acc0[r] + Vx[vxiB[r] + g] + Vx[vxjB[r] + g];
          out[(size_t)(Mbase + half * 8 + r) * Hh + g] = v;
        }
      }
    }
    // epilogue tile 1 (g in [32*ntp+16, 32*ntp+32)) — tile 19 fully masked
    {
      const int g = ntp * 32 + 16 + nl;
      if (g < Hh) {
#pragma unroll
        for (int r = 0; r < 8; ++r) {
          const float v = acc1[r] + Vx[vxiB[r] + g] + Vx[vxjB[r] + g];
          out[(size_t)(Mbase + half * 8 + r) * Hh + g] = v;
        }
      }
    }
  }
}

// ---------------------------------------------------------------------------
extern "C" void kernel_launch(void* const* d_in, const int* in_sizes, int n_in,
                              void* d_out, int out_size, void* d_ws, size_t ws_size,
                              hipStream_t stream) {
  const float* x   = (const float*)d_in[0];
  const float* e   = (const float*)d_in[1];
  const float* U_w = (const float*)d_in[2];
  const float* U_b = (const float*)d_in[3];
  const float* V_w = (const float*)d_in[4];
  const float* V_b = (const float*)d_in[5];
  float*       out = (float*)d_out;
  float*       Vx  = (float*)d_ws;  // B*N*H floats = 960000 bytes of scratch

  vx_bias_kernel<<<Bb * Nn, 128, 0, stream>>>(x, V_w, V_b, U_b, Vx);

  const int    tiles  = (Bb * Nn * Nn) / MT;  // 10000
  const int    blocks = tiles / WAVES;        // 1250
  const size_t lds    = (size_t)(GP + WAVES * MT) * KSTRIDE * sizeof(__bf16);  // 293888 B
  edge_wmma_kernel<<<blocks, WAVES * 32, lds, stream>>>(e, U_w, Vx, out);
}
